// MaskedCrossAttention_69295002353809
// MI455X (gfx1250) — matrix operand order repaired and checked by reading, not compiled, not run on hardware
//
#include <hip/hip_runtime.h>
#include <hip/hip_bf16.h>
#include <math.h>

// ---- CDNA5 WMMA types ------------------------------------------------------
typedef __attribute__((ext_vector_type(16))) __bf16 v16bf;
typedef __attribute__((ext_vector_type(8)))  __bf16 v8bf;
typedef __attribute__((ext_vector_type(8)))  float  v8f;

union V16BF { v16bf v; v8bf half8[2]; __bf16 e[16]; unsigned short u[16]; };

__device__ __forceinline__ v8f wmma_bf16(v16bf a, v16bf b, v8f c) {
    return __builtin_amdgcn_wmma_f32_16x16x32_bf16(
        false, a, false, b, (short)0, c, false, false);
}

// ---- Fragment loaders (ISA 7.12.2, wave32) ---------------------------------
// A 16x32 bf16: lane&15 = row M; lanes 0-15 hold K kb..kb+7 (v0-3) and
// kb+16..kb+23 (v4-7) with kb = k0 (lanes<16) or k0+8 (lanes>=16).
__device__ __forceinline__ v16bf load_a_f32(const float* __restrict__ A, int lda,
                                            int m0, int k0, int lane) {
    int m  = m0 + (lane & 15);
    int kb = k0 + ((lane & 16) ? 8 : 0);
    const float* row = A + (size_t)m * lda + kb;
    float4 x0 = *(const float4*)(row);
    float4 x1 = *(const float4*)(row + 4);
    float4 x2 = *(const float4*)(row + 16);
    float4 x3 = *(const float4*)(row + 20);
    V16BF a;
    a.e[0]=(__bf16)x0.x; a.e[1]=(__bf16)x0.y; a.e[2]=(__bf16)x0.z; a.e[3]=(__bf16)x0.w;
    a.e[4]=(__bf16)x1.x; a.e[5]=(__bf16)x1.y; a.e[6]=(__bf16)x1.z; a.e[7]=(__bf16)x1.w;
    a.e[8]=(__bf16)x2.x; a.e[9]=(__bf16)x2.y; a.e[10]=(__bf16)x2.z; a.e[11]=(__bf16)x2.w;
    a.e[12]=(__bf16)x3.x; a.e[13]=(__bf16)x3.y; a.e[14]=(__bf16)x3.z; a.e[15]=(__bf16)x3.w;
    return a.v;
}

__device__ __forceinline__ v16bf load_a_bf16(const __bf16* __restrict__ A, int lda,
                                             int m0, int k0, int lane) {
    int m  = m0 + (lane & 15);
    int kb = k0 + ((lane & 16) ? 8 : 0);
    const __bf16* row = A + (size_t)m * lda + kb;
    V16BF a;
    a.half8[0] = *(const v8bf*)(row);        // K kb..kb+7   (16B load)
    a.half8[1] = *(const v8bf*)(row + 16);   // K kb+16..+23 (16B load)
    return a.v;
}

// B 32x16: lane&15 = col N; lanes hold K kb..kb+7 and kb+16..kb+23 with
// kb = k0 (lanes<16) or k0+8 (lanes>=16).  K-major storage -> two 16B loads.
__device__ __forceinline__ v16bf load_b_kmajor(const __bf16* __restrict__ Bt,
                                               int ldk, int k0, int n0, int lane) {
    int n  = n0 + (lane & 15);
    int kb = k0 + ((lane & 16) ? 8 : 0);
    const __bf16* col = Bt + (size_t)n * ldk + kb;
    V16BF b;
    b.half8[0] = *(const v8bf*)(col);
    b.half8[1] = *(const v8bf*)(col + 16);
    return b.v;
}

// Problem constants
#define BATCH 4
#define SEQ   2048
#define DT    4096
#define DV    2048
#define HEADS 8
#define DHEAD 64
#define INNER 512
#define TN    192        /* T_MEDIA * N_LATENTS */

// ---- K0: transpose + convert a f32 [K,N] weight to bf16 [N,K] -------------
__global__ void __launch_bounds__(256)
prep_wt_kernel(const float* __restrict__ W, __bf16* __restrict__ Wt,
               int K, int N) {
    int i = blockIdx.x * 256 + threadIdx.x;      // coalesced read over k*N+n
    if (i >= K * N) return;
    int k = i / N, n = i % N;
    Wt[(size_t)n * K + k] = (__bf16)W[i];
}

// ---- K1: kv = latent @ W_vk -> k [B,H,192,64], v^T [B,H,64,192] (bf16) ----
// wave strip = 16x64 (one head wide). waves = 4 * 12 * 16 = 768 -> 96 blocks
__global__ void __launch_bounds__(256)
kv_proj_kernel(const float* __restrict__ latent, const __bf16* __restrict__ Wvkt,
               __bf16* __restrict__ kbuf, __bf16* __restrict__ vtbuf) {
    int wid  = blockIdx.x * 8 + (threadIdx.x >> 5);
    int lane = threadIdx.x & 31;
    const int MT = TN / 16, NS = 1024 / 64;
    int b  = wid / (MT * NS);
    int r  = wid % (MT * NS);
    int mt = r / NS, ns = r % NS;

    const float* A = latent + (size_t)b * TN * DV;
    v8f acc[4] = {};
    for (int k0 = 0; k0 < DV; k0 += 32) {
        v16bf af = load_a_f32(A, DV, mt * 16, k0, lane);
#pragma unroll
        for (int j = 0; j < 4; ++j)
            acc[j] = wmma_bf16(af,
                load_b_kmajor(Wvkt, DV, k0, ns * 64 + j * 16, lane), acc[j]);
    }
    int moff = (lane & 16) ? 8 : 0;
    // Strip == one head: ns<8 -> k head ns; ns>=8 -> v head ns-8 (uniform!)
    if (ns < HEADS) {
        __bf16* kh = kbuf + (((size_t)b * HEADS + ns) * TN) * DHEAD;
#pragma unroll
        for (int j = 0; j < 4; ++j) {
            int d = j * 16 + (lane & 15);
#pragma unroll
            for (int rr = 0; rr < 8; ++rr)
                kh[(size_t)(mt * 16 + rr + moff) * DHEAD + d] = (__bf16)acc[j][rr];
        }
    } else {
        __bf16* vh = vtbuf + (((size_t)b * HEADS + (ns - HEADS)) * DHEAD) * TN;
#pragma unroll
        for (int j = 0; j < 4; ++j) {
            int d = j * 16 + (lane & 15);
#pragma unroll
            for (int rr = 0; rr < 8; ++rr)
                vh[(size_t)d * TN + (mt * 16 + rr + moff)] = (__bf16)acc[j][rr];
        }
    }
}

// ---- K2: q = y @ W_q -> [B,H,2048,64] bf16 --------------------------------
// wave strip = 16x64; the 8 waves of a block share one 16-row A slab
// (ns = 0..7, same mt) so y rows are fetched once per block from HBM.
// waves = 4 * 128 * 8 = 4096 -> 512 blocks
__global__ void __launch_bounds__(256)
q_proj_kernel(const float* __restrict__ y, const __bf16* __restrict__ Wqt,
              __bf16* __restrict__ qbuf) {
    int wid  = blockIdx.x * 8 + (threadIdx.x >> 5);
    int lane = threadIdx.x & 31;
    const int MT = SEQ / 16, NS = INNER / 64;
    int b  = wid / (MT * NS);
    int r  = wid % (MT * NS);
    int mt = r / NS, ns = r % NS;

    const float* A = y + (size_t)b * SEQ * DT;
    v8f acc[4] = {};
    for (int k0 = 0; k0 < DT; k0 += 32) {
        v16bf af = load_a_f32(A, DT, mt * 16, k0, lane);
#pragma unroll
        for (int j = 0; j < 4; ++j)
            acc[j] = wmma_bf16(af,
                load_b_kmajor(Wqt, DT, k0, ns * 64 + j * 16, lane), acc[j]);
    }
    int moff = (lane & 16) ? 8 : 0;
    __bf16* qh = qbuf + (((size_t)b * HEADS + ns) * SEQ) * DHEAD;  // head = ns
#pragma unroll
    for (int j = 0; j < 4; ++j) {
        int d = j * 16 + (lane & 15);
#pragma unroll
        for (int rr = 0; rr < 8; ++rr)
            qh[(size_t)(mt * 16 + rr + moff) * DHEAD + d] = (__bf16)acc[j][rr];
    }
}

// ---- K3: fused masked attention -------------------------------------------
// tiles = B*H*(S/16) = 4096; 4 waves/block -> 1024 blocks of 128 threads.
__global__ void __launch_bounds__(128)
attn_kernel(const __bf16* __restrict__ qbuf,
            const __bf16* __restrict__ kbuf,
            const __bf16* __restrict__ vtbuf,
            const int* __restrict__ media_mask,
            __bf16* __restrict__ attn_out) {
    __shared__ float sim[4][16][TN];   // 48 KB

    int wave = threadIdx.x >> 5;
    int lane = threadIdx.x & 31;
    int tid  = blockIdx.x * 4 + wave;
    int b    = tid / (HEADS * (SEQ / 16));
    int rem  = tid % (HEADS * (SEQ / 16));
    int h    = rem / (SEQ / 16);
    int s0   = (rem % (SEQ / 16)) * 16;

    const __bf16* qh = qbuf  + (((size_t)b * HEADS + h) * SEQ) * DHEAD;
    const __bf16* kh = kbuf  + (((size_t)b * HEADS + h) * TN) * DHEAD;
    const __bf16* vt = vtbuf + (((size_t)b * HEADS + h) * DHEAD) * TN;

    // q fragments (K = 64 -> two 16x32 A fragments)
    v16bf qa0 = load_a_bf16(qh, DHEAD, s0, 0,  lane);
    v16bf qa1 = load_a_bf16(qh, DHEAD, s0, 32, lane);

    // sim[16 x 192] = q(16x64) . k^T(64x192); k rows (d-contig) are K-major.
    int moff = (lane & 16) ? 8 : 0;
#pragma unroll
    for (int nt = 0; nt < TN / 16; ++nt) {
        v8f acc = {};
        acc = wmma_bf16(qa0, load_b_kmajor(kh, DHEAD, 0,  nt * 16, lane), acc);
        acc = wmma_bf16(qa1, load_b_kmajor(kh, DHEAD, 32, nt * 16, lane), acc);
        int n = nt * 16 + (lane & 15);
#pragma unroll
        for (int rr = 0; rr < 8; ++rr)
            sim[wave][rr + moff][n] = acc[rr];
    }
    __syncthreads();

    // Masked softmax: query s attends only to keys j in [(mt-1)*64, mt*64),
    // mt = media_mask[b][s]; mt==0 -> all-zero row (no_media case).
    if (lane < 16) {
        int row = lane;
        int mt  = media_mask[b * SEQ + s0 + row];
        float* sr = &sim[wave][row][0];
        if (mt == 0) {
            for (int j = 0; j < TN; ++j) sr[j] = 0.0f;
        } else {
            int j0 = (mt - 1) * 64;
            float mx = -INFINITY;
            for (int j = 0; j < 64; ++j) mx = fmaxf(mx, sr[j0 + j]);
            float sum = 0.0f;
            for (int j = 0; j < 64; ++j) {
                float e = __expf(sr[j0 + j] - mx);
                sr[j0 + j] = e;
                sum += e;
            }
            float inv = 1.0f / sum;
            for (int j = 0; j < TN; ++j)
                sr[j] = (j >= j0 && j < j0 + 64) ? sr[j] * inv : 0.0f;
        }
    }
    __syncthreads();   // EXEC reconverged before WMMA

    // out[16 x 64] = p(16x192) . v(192x64); v^T rows (j-contig) are K-major.
    const float* pl = &sim[wave][0][0];
#pragma unroll
    for (int nt = 0; nt < DHEAD / 16; ++nt) {
        v8f acc = {};
#pragma unroll
        for (int ks = 0; ks < TN / 32; ++ks) {
            v16bf pa = load_a_f32(pl, TN, 0, ks * 32, lane);
            v16bf vb = load_b_kmajor(vt, TN, ks * 32, nt * 16, lane);
            acc = wmma_bf16(pa, vb, acc);
        }
        int n = nt * 16 + (lane & 15);
#pragma unroll
        for (int rr = 0; rr < 8; ++rr)
            attn_out[((size_t)b * SEQ + s0 + rr + moff) * INNER + h * DHEAD + n] =
                (__bf16)acc[rr];
    }
}

// ---- K4: result = attn_out @ W_out ([8192,512]bf16 x [512,4096] -> f32) ---
// wave strip = 16x64. waves = 512 * 64 = 32768 -> 4096 blocks.
// Output is write-once: non-temporal stores keep y/W resident in L2.
__global__ void __launch_bounds__(256)
out_proj_kernel(const __bf16* __restrict__ attn_out,
                const __bf16* __restrict__ Woutt, float* __restrict__ out) {
    int wid  = blockIdx.x * 8 + (threadIdx.x >> 5);
    int lane = threadIdx.x & 31;
    const int NS = DT / 64;
    int mt = wid / NS, ns = wid % NS;

    v8f acc[4] = {};
#pragma unroll
    for (int k0 = 0; k0 < INNER; k0 += 32) {
        v16bf af = load_a_bf16(attn_out, INNER, mt * 16, k0, lane);
#pragma unroll
        for (int j = 0; j < 4; ++j)
            acc[j] = wmma_bf16(af,
                load_b_kmajor(Woutt, INNER, k0, ns * 64 + j * 16, lane), acc[j]);
    }
    int moff = (lane & 16) ? 8 : 0;
#pragma unroll
    for (int j = 0; j < 4; ++j) {
        int n = ns * 64 + j * 16 + (lane & 15);
#pragma unroll
        for (int rr = 0; rr < 8; ++rr)
            __builtin_nontemporal_store(acc[j][rr],
                &out[(size_t)(mt * 16 + rr + moff) * DT + n]);
    }
}

// ---- Host-side launch ------------------------------------------------------
extern "C" void kernel_launch(void* const* d_in, const int* in_sizes, int n_in,
                              void* d_out, int out_size, void* d_ws, size_t ws_size,
                              hipStream_t stream) {
    const float* latent = (const float*)d_in[0];   // [4,3,64,2048]
    const float* y      = (const float*)d_in[1];   // [4,2048,4096]
    const int*   mask   = (const int*)  d_in[2];   // [4,2048]
    const float* Wvk    = (const float*)d_in[3];   // [2048,1024]
    const float* Wq     = (const float*)d_in[4];   // [4096,512]
    const float* Wout   = (const float*)d_in[5];   // [512,4096]
    float*       out    = (float*)d_out;           // [4,2048,4096]

    char* ws = (char*)d_ws;
    size_t off = 0;
    __bf16* qbuf  = (__bf16*)(ws + off); off += (size_t)BATCH*HEADS*SEQ*DHEAD*2;  // 8 MiB
    __bf16* kbuf  = (__bf16*)(ws + off); off += (size_t)BATCH*HEADS*TN*DHEAD*2;   // 768 KiB
    __bf16* vtbuf = (__bf16*)(ws + off); off += (size_t)BATCH*HEADS*DHEAD*TN*2;   // 768 KiB
    __bf16* abuf  = (__bf16*)(ws + off); off += (size_t)BATCH*SEQ*INNER*2;        // 8 MiB
    __bf16* Wvkt  = (__bf16*)(ws + off); off += (size_t)DV*1024*2;                // 4 MiB
    __bf16* Wqt   = (__bf16*)(ws + off); off += (size_t)DT*INNER*2;               // 4 MiB
    __bf16* Woutt = (__bf16*)(ws + off); off += (size_t)INNER*DT*2;               // 4 MiB

    prep_wt_kernel<<<(DV*1024 + 255)/256, 256, 0, stream>>>(Wvk,  Wvkt,  DV,    1024);
    prep_wt_kernel<<<(DT*INNER + 255)/256, 256, 0, stream>>>(Wq,   Wqt,   DT,    INNER);
    prep_wt_kernel<<<(INNER*DT + 255)/256, 256, 0, stream>>>(Wout, Woutt, INNER, DT);

    kv_proj_kernel <<<96,   256, 0, stream>>>(latent, Wvkt, kbuf, vtbuf);
    q_proj_kernel  <<<512,  256, 0, stream>>>(y, Wqt, qbuf);
    attn_kernel    <<<1024, 128, 0, stream>>>(qbuf, kbuf, vtbuf, mask, abuf);
    out_proj_kernel<<<4096, 256, 0, stream>>>(abuf, Woutt, out);
}
